// GNNLayer_84628035601116
// MI455X (gfx1250) — matrix operand order repaired and checked by reading, not compiled, not run on hardware
//
#include <hip/hip_runtime.h>
#include <hip/hip_bf16.h>

typedef __attribute__((ext_vector_type(16))) __bf16 v16bf;
typedef __attribute__((ext_vector_type(8)))  float  v8f;
typedef __attribute__((ext_vector_type(4)))  float  f32x4;

#define NNODE 50000
#define NEDGE 800000
#define MT    3125   /* 50000 / 16 : exact, no edge masking needed */

union ABFrag { v16bf v; f32x4 q[2]; };

// ---------------- fp32 -> bf16 conversion ----------------
__global__ void k_cvt_bf16(const float* __restrict__ x, __bf16* __restrict__ o, int n) {
    int t = blockIdx.x * blockDim.x + threadIdx.x;
    if (t < n) o[t] = (__bf16)x[t];
}

// ---------------- pack W[K][NN] (row-major fp32) into WMMA B-fragment order ----------------
// Bp[kt][half][n][j] = W[32*kt + 16*half + j][n]   (bf16)
__global__ void k_pack_w(const float* __restrict__ W, __bf16* __restrict__ Wp,
                         int NN, int total) {
    int t = blockIdx.x * blockDim.x + threadIdx.x;
    if (t >= total) return;
    int j = t & 15;
    int r = t >> 4;
    int n = r % NN; r /= NN;
    int half = r & 1;
    int kt = r >> 1;
    int k = kt * 32 + half * 16 + j;
    Wp[t] = (__bf16)W[k * NN + n];
}

// ---------------- bf16 WMMA GEMM: C[M,NN] = A[M,256] * B[256,NN] + bias ----------------
// Block = 4 waves. The block's 64-column B tile (8 kt x 2 half x 64 col x 16 bf16 = 32KB)
// is staged into LDS once with async global->LDS DMA (ASYNCcnt), then every wave's
// B fragments come from ds_load_b128 while A streams from global. One wave computes
// a 16x64 output tile (4 f32 accumulators), K=256 in 8 WMMA steps.
template <int NN, bool RELU_BF16>
__global__ void k_gemm_wmma(const __bf16* __restrict__ A,
                            const __bf16* __restrict__ Bp,
                            const float* __restrict__ bias,
                            void* __restrict__ Cout) {
    __shared__ __align__(16) __bf16 Bs[8 * 2 * 64 * 16];   // 32 KB

    const int tid   = threadIdx.x;
    const int lane  = tid & 31;
    const int wave  = tid >> 5;
    const int mt    = blockIdx.x * 4 + wave;
    const int nbase = blockIdx.y * 64;

    // ---- stage B tile: 2048 x 16B chunks, 16 per thread, async to LDS ----
    {
        const unsigned lbase = (unsigned)(uintptr_t)(&Bs[0]);
#pragma unroll
        for (int i = 0; i < 16; ++i) {
            int c    = tid + i * 128;
            int kt   = c >> 8;
            int rem  = c & 255;
            int half = rem >> 7;
            int n    = (rem >> 1) & 63;
            int q    = rem & 1;
            const __bf16* gsrc = Bp + ((size_t)(kt * 2 + half) * NN + nbase + n) * 16 + q * 8;
            unsigned ldst = lbase + (unsigned)c * 16;
            asm volatile("global_load_async_to_lds_b128 %0, %1, off"
                         :: "v"(ldst), "v"((unsigned long long)(uintptr_t)gsrc)
                         : "memory");
        }
        asm volatile("s_wait_asynccnt 0x0" ::: "memory");
    }
    __syncthreads();

    if (mt >= MT) return;                 // wave-uniform exit: EXEC all-1s for WMMA
    const int half = lane >> 4;
    const int l16  = lane & 15;
    const int arow = mt * 16 + l16;       // A: lanes 0-15 and 16-31 both hold M = lane%16

    v8f acc[4];
#pragma unroll
    for (int a = 0; a < 4; a++) {
        float bv = bias[nbase + a * 16 + l16];   // bias depends on column only
        v8f c;
#pragma unroll
        for (int r = 0; r < 8; r++) c[r] = bv;
        acc[a] = c;
    }

    const f32x4* Ar = (const f32x4*)(A + (size_t)arow * 256);
#pragma unroll
    for (int kt = 0; kt < 8; ++kt) {
        // A fragment: elems 0..15 = K[kt*32+16*half .. +15]
        ABFrag af;
        int c0 = (kt * 32 + half * 16) >> 3;   // f32x4 (8 x bf16) chunk index
        af.q[0] = Ar[c0];
        af.q[1] = Ar[c0 + 1];
#pragma unroll
        for (int a = 0; a < 4; a++) {
            const f32x4* bsp = (const f32x4*)(Bs +
                ((size_t)(kt * 2 + half) * 64 + a * 16 + l16) * 16);
            ABFrag bfr;
            bfr.q[0] = bsp[0];
            bfr.q[1] = bsp[1];
            acc[a] = __builtin_amdgcn_wmma_f32_16x16x32_bf16(
                false, af.v, false, bfr.v, (short)0, acc[a], false, false);
        }
    }

#pragma unroll
    for (int a = 0; a < 4; a++) {
        int col = nbase + a * 16 + l16;
#pragma unroll
        for (int r = 0; r < 8; r++) {
            int row = mt * 16 + r + half * 8;   // C/D: lanes 16-31 hold M = r+8
            float v = acc[a][r];
            if (RELU_BF16) {
                v = fmaxf(v, 0.0f);
                ((__bf16*)Cout)[(size_t)row * NN + col] = (__bf16)v;
            } else {
                ((float*)Cout)[(size_t)row * NN + col] = v;
            }
        }
    }
}

// ---------------- per-node attention dot products: alpha = f.aw[:128], beta = f.aw[128:256] ----------------
__global__ void k_attn_ab(const float* __restrict__ f, const float* __restrict__ aw,
                          float* __restrict__ al, float* __restrict__ be) {
    int gw   = (blockIdx.x * blockDim.x + threadIdx.x) >> 5;
    int lane = threadIdx.x & 31;
    if (gw >= NNODE) return;
    f32x4 fv = *(const f32x4*)(f + (size_t)gw * 128 + lane * 4);
    float a0 = 0.f, b0 = 0.f;
#pragma unroll
    for (int c = 0; c < 4; c++) {
        a0 += fv[c] * aw[lane * 4 + c];
        b0 += fv[c] * aw[128 + lane * 4 + c];
    }
#pragma unroll
    for (int off = 16; off > 0; off >>= 1) {
        a0 += __shfl_xor(a0, off, 32);
        b0 += __shfl_xor(b0, off, 32);
    }
    if (lane == 0) { al[gw] = a0; be[gw] = b0; }
}

__global__ void k_zero(float* __restrict__ p, int n) {
    int t = blockIdx.x * blockDim.x + threadIdx.x;
    if (t < n) p[t] = 0.f;
}

// ---------------- edge scatter: one wave per edge ----------------
// acc[n][0:128] += w * f[dst], acc[n][128] += w*elem, acc[n][129] += w
__global__ void k_edge(const float* __restrict__ f, const float* __restrict__ al,
                       const float* __restrict__ be, const float* __restrict__ elem,
                       const int* __restrict__ idx0, const int* __restrict__ idx1,
                       const float* __restrict__ awt, const float* __restrict__ atb,
                       float* __restrict__ acc) {
    int gw   = (blockIdx.x * blockDim.x + threadIdx.x) >> 5;
    int lane = threadIdx.x & 31;
    if (gw >= NEDGE) return;
    int s = idx0[gw];
    int d = idx1[gw];
    float el    = elem[gw];
    float logit = al[s] + be[d] + el * awt[0] + atb[0];
    float t  = logit * 0.05f;                 // logits / 20
    float lr = t > 0.f ? t : 0.2f * t;        // leaky_relu, slope 0.2
    float w  = __expf(-lr);
    f32x4 fd = *(const f32x4*)(f + (size_t)d * 128 + lane * 4);
    float* ap = acc + (size_t)s * 130 + lane * 4;
    atomicAdd(ap + 0, w * fd[0]);
    atomicAdd(ap + 1, w * fd[1]);
    atomicAdd(ap + 2, w * fd[2]);
    atomicAdd(ap + 3, w * fd[3]);
    if (lane == 0) {
        atomicAdd(acc + (size_t)s * 130 + 128, w * el);
        atomicAdd(acc + (size_t)s * 130 + 129, w);
    }
}

// ---------------- assemble output: out[n, h*257 + j] ----------------
// j<128: f[n][j] (src half normalizes to f exactly); 128..255: acc/rs; 256: acc_elem/rs
__global__ void k_final(const float* __restrict__ f, const float* __restrict__ acc,
                        float* __restrict__ out, int h) {
    int t = blockIdx.x * blockDim.x + threadIdx.x;
    if (t >= NNODE * 257) return;
    int n = t / 257;
    int j = t - n * 257;
    float rs = acc[(size_t)n * 130 + 129];
    float v;
    if (j < 128)      v = f[(size_t)n * 128 + j];
    else if (j < 256) v = acc[(size_t)n * 130 + (j - 128)] / rs;
    else              v = acc[(size_t)n * 130 + 128] / rs;
    out[(size_t)n * 514 + h * 257 + j] = v;
}

extern "C" void kernel_launch(void* const* d_in, const int* in_sizes, int n_in,
                              void* d_out, int out_size, void* d_ws, size_t ws_size,
                              hipStream_t stream) {
    const float* x    = (const float*)d_in[0];
    const float* elem = (const float*)d_in[1];
    const float* W1   = (const float*)d_in[2];
    const float* b1   = (const float*)d_in[3];
    const float* W2   = (const float*)d_in[4];
    const float* b2   = (const float*)d_in[5];
    const float* aw   = (const float*)d_in[6];
    const float* ab   = (const float*)d_in[7];
    const int*   idx  = (const int*)d_in[8];
    float* out = (float*)d_out;
    (void)in_sizes; (void)n_in; (void)out_size; (void)ws_size;

    size_t off = 0;
    auto wsalloc = [&](size_t bytes) -> char* {
        char* p = (char*)d_ws + off;
        off += (bytes + 255) & ~(size_t)255;
        return p;
    };
    __bf16* xbf = (__bf16*)wsalloc((size_t)NNODE * 256 * 2);
    __bf16* w1p = (__bf16*)wsalloc((size_t)256 * 256 * 2);
    __bf16* w2p = (__bf16*)wsalloc((size_t)256 * 128 * 2);
    __bf16* h1  = (__bf16*)wsalloc((size_t)NNODE * 256 * 2);
    float*  f   = (float*) wsalloc((size_t)NNODE * 128 * 4);
    float*  al  = (float*) wsalloc((size_t)NNODE * 4);
    float*  be  = (float*) wsalloc((size_t)NNODE * 4);
    float*  acc = (float*) wsalloc((size_t)NNODE * 130 * 4);

    {
        int n = NNODE * 256;
        k_cvt_bf16<<<(n + 255) / 256, 256, 0, stream>>>(x, xbf, n);
    }

    for (int h = 0; h < 2; ++h) {
        k_pack_w<<<(256 * 256 + 255) / 256, 256, 0, stream>>>(W1 + h * 256 * 256, w1p, 256, 256 * 256);
        k_pack_w<<<(256 * 128 + 255) / 256, 256, 0, stream>>>(W2 + h * 256 * 128, w2p, 128, 256 * 128);

        // h1 = relu(x @ W1 + b1)  : [N,256] bf16
        k_gemm_wmma<256, true ><<<dim3((MT + 3) / 4, 4), 128, 0, stream>>>(xbf, w1p, b1 + h * 256, h1);
        // f  = h1 @ W2 + b2       : [N,128] fp32
        k_gemm_wmma<128, false><<<dim3((MT + 3) / 4, 2), 128, 0, stream>>>(h1, w2p, b2 + h * 128, f);

        k_attn_ab<<<(NNODE + 7) / 8, 256, 0, stream>>>(f, aw + h * 257, al, be);

        {
            int n = NNODE * 130;
            k_zero<<<(n + 255) / 256, 256, 0, stream>>>(acc, n);
        }

        k_edge<<<(NEDGE + 7) / 8, 256, 0, stream>>>(f, al, be, elem,
                                                    idx, idx + NEDGE,
                                                    aw + h * 257 + 256, ab + h, acc);

        {
            int n = NNODE * 257;
            k_final<<<(n + 255) / 256, 256, 0, stream>>>(f, acc, out, h);
        }
    }
}
